// Network_65335042507166
// MI455X (gfx1250) — matrix operand order repaired and checked by reading, not compiled, not run on hardware
//
#include <hip/hip_runtime.h>
#include <cstdint>
#include <cstddef>

// Problem constants (match reference)
#define B_   64
#define S_   512
#define H_   768
#define N_   16
#define P_   240      // N*(N-1)
#define CC_  5
#define CR_  3
#define HID_ 512      // classifier hidden width

typedef __attribute__((ext_vector_type(16))) _Float16 v16h;
typedef __attribute__((ext_vector_type(8)))  _Float16 v8h;
typedef __attribute__((ext_vector_type(8)))  float    v8f;
typedef __attribute__((ext_vector_type(4)))  uint32_t u32x4;

union AFrag { v16h v; v8h h[2]; };
union BFrag { v16h v; u32x4 q[2]; };

__device__ __forceinline__ void loadA(AFrag& a, const _Float16* p) {
    a.h[0] = *(const v8h*)(p);         // K = half*8 + 0..7
    a.h[1] = *(const v8h*)(p + 16);    // K = 16 + half*8 + 0..7
}
__device__ __forceinline__ void loadB(BFrag& b, const uint32_t* p) {
    b.q[0] = *(const u32x4*)(p);
    b.q[1] = *(const u32x4*)(p + 4);
}
__device__ __forceinline__ v8f wmma(const AFrag& a, const BFrag& b, v8f c) {
    return __builtin_amdgcn_wmma_f32_16x16x32_f16(false, a.v, false, b.v,
                                                  (short)0, c, false, false);
}

// ---------------------------------------------------------------------------
// Pack f32 weight matrix W[K, Ncols] (row-major, rows offset by rowOff) into
// a WMMA-native tiled layout:
//   dst[ ((tn * (K/32) + ks) * 32 + lane) * 8 + w ] =
//       pack_f16( W[rowOff + 2*kp][n], W[rowOff + 2*kp + 1][n] )
//   with n  = tn*16 + (lane&15),  kp = ks*16 + (w>>2)*8 + (lane>>4)*4 + (w&3)
// so in the GEMM each lane reads its whole 8-dword B fragment with two
// contiguous b128 loads and a constant per-kstep pointer bump.
// ---------------------------------------------------------------------------
__global__ void pack_b_tiles(const float* __restrict__ src, uint32_t* __restrict__ dst,
                             int Ncols, int K, int rowOff) {
    int gid = blockIdx.x * blockDim.x + threadIdx.x;
    int ksteps = K >> 5;
    int total = (Ncols >> 4) * ksteps * 256;   // 32 lanes * 8 dwords per (tn,ks)
    if (gid >= total) return;
    int w    = gid & 7;
    int lane = (gid >> 3) & 31;
    int rest = gid >> 8;
    int ks   = rest % ksteps;
    int tn   = rest / ksteps;
    int n    = tn * 16 + (lane & 15);
    int kp   = ks * 16 + (w >> 2) * 8 + (lane >> 4) * 4 + (w & 3);
    float f0 = src[(size_t)(rowOff + 2 * kp) * Ncols + n];
    float f1 = src[(size_t)(rowOff + 2 * kp + 1) * Ncols + n];
    union { _Float16 h[2]; uint32_t u; } pk;
    pk.h[0] = (_Float16)f0;
    pk.h[1] = (_Float16)f1;
    dst[gid] = pk.u;
}

// ---------------------------------------------------------------------------
// Span mean pooling: one block per (b,n) span; 256 threads stride H columns.
// ---------------------------------------------------------------------------
__global__ void pool_spans(const float* __restrict__ lhs, const int* __restrict__ starts,
                           const int* __restrict__ lens, float* __restrict__ pooled,
                           _Float16* __restrict__ pooledH) {
    int span = blockIdx.x;          // b*N + n
    int b  = span >> 4;
    int st = starts[span];
    int ln = lens[span];
    float inv = 1.0f / (float)(ln + 1);
    const float* base = lhs + ((size_t)b * S_ + st) * H_;
    for (int h = threadIdx.x; h < H_; h += 256) {
        float acc = 0.f;
        const float* p = base + h;
        for (int s = 0; s <= ln; ++s) acc += p[(size_t)s * H_];
        float val = acc * inv;
        pooled[(size_t)span * H_ + h]  = val;
        pooledH[(size_t)span * H_ + h] = (_Float16)val;
    }
}

// ---------------------------------------------------------------------------
// WMMA GEMM, 16x64 output strip per wave (4 N-tiles, 4 accumulators), with
// explicit 2-deep software pipeline (ping-pong register sets) so next-step
// b128 loads overlap the current step's 4 WMMAs.
// __launch_bounds__(32,1): lift the VGPR cap so both pipeline register sets
// (32 C + 16 A + 64 B + addresses) stay live; latency is hidden by the
// thousands of independent single-wave workgroups, not by per-SIMD waves.
// Requires ksteps even and >= 2 (true for all K=768 GEMMs here).
// ---------------------------------------------------------------------------
__global__ void __launch_bounds__(32, 1)
wmma_gemm4(const _Float16* __restrict__ A, const uint32_t* __restrict__ Bp,
           const float* __restrict__ bias, float* __restrict__ C,
           int Ncols, int K, int actTanh) {
    const int lane = threadIdx.x;    // blockDim.x == 32 (one wave)
    const int half = lane >> 4;
    const int l16  = lane & 15;
    const int tm   = blockIdx.x;     // M tile (16 rows)
    const int tg   = blockIdx.y;     // group of 4 N tiles (64 cols)
    const int ksteps = K >> 5;

    const _Float16* ap = A + (size_t)(tm * 16 + l16) * K + half * 8;
    const uint32_t* bp0 = Bp + ((size_t)((tg * 4 + 0) * ksteps) * 32 + lane) * 8;
    const uint32_t* bp1 = Bp + ((size_t)((tg * 4 + 1) * ksteps) * 32 + lane) * 8;
    const uint32_t* bp2 = Bp + ((size_t)((tg * 4 + 2) * ksteps) * 32 + lane) * 8;
    const uint32_t* bp3 = Bp + ((size_t)((tg * 4 + 3) * ksteps) * 32 + lane) * 8;

    v8f c0 = {}, c1 = {}, c2 = {}, c3 = {};

    AFrag a0, a1;
    BFrag b00, b01, b02, b03, b10, b11, b12, b13;

    // prologue: step 0 into set 0
    loadA(a0, ap);
    loadB(b00, bp0); loadB(b01, bp1); loadB(b02, bp2); loadB(b03, bp3);

    // steady state: pointers always reference the step held in set 0
    for (int ks = 0; ks + 2 < ksteps; ks += 2) {
        // prefetch step ks+1 into set 1
        loadA(a1, ap + 32);
        loadB(b10, bp0 + 256); loadB(b11, bp1 + 256);
        loadB(b12, bp2 + 256); loadB(b13, bp3 + 256);
        // consume step ks
        c0 = wmma(a0, b00, c0); c1 = wmma(a0, b01, c1);
        c2 = wmma(a0, b02, c2); c3 = wmma(a0, b03, c3);
        ap += 64; bp0 += 512; bp1 += 512; bp2 += 512; bp3 += 512;
        // prefetch step ks+2 into set 0
        loadA(a0, ap);
        loadB(b00, bp0); loadB(b01, bp1); loadB(b02, bp2); loadB(b03, bp3);
        // consume step ks+1
        c0 = wmma(a1, b10, c0); c1 = wmma(a1, b11, c1);
        c2 = wmma(a1, b12, c2); c3 = wmma(a1, b13, c3);
    }
    // epilogue: steps ksteps-2 (set 0, already loaded) and ksteps-1
    loadA(a1, ap + 32);
    loadB(b10, bp0 + 256); loadB(b11, bp1 + 256);
    loadB(b12, bp2 + 256); loadB(b13, bp3 + 256);
    c0 = wmma(a0, b00, c0); c1 = wmma(a0, b01, c1);
    c2 = wmma(a0, b02, c2); c3 = wmma(a0, b03, c3);
    c0 = wmma(a1, b10, c0); c1 = wmma(a1, b11, c1);
    c2 = wmma(a1, b12, c2); c3 = wmma(a1, b13, c3);

    // D layout: VGPR r -> row M = tm*16 + half*8 + r, col = tn*16 + l16.
    const int rowBase = tm * 16 + half * 8;
    v8f cc[4] = {c0, c1, c2, c3};
    #pragma unroll
    for (int s = 0; s < 4; ++s) {
        int nc = (tg * 4 + s) * 16 + l16;
        float bv = bias ? bias[nc] : 0.0f;
        #pragma unroll
        for (int r = 0; r < 8; ++r) {
            float x = cc[s][r] + bv;
            if (actTanh) x = tanhf(x);
            C[(size_t)(rowBase + r) * Ncols + nc] = x;
        }
    }
}

// ---------------------------------------------------------------------------
// Per-batch 16x16 attention + residual + layernorm (f32 for accuracy).
// One block (256 threads) per batch element.
// ---------------------------------------------------------------------------
__global__ void attn_ln(const float* __restrict__ pooled,
                        const float* __restrict__ q, const float* __restrict__ k,
                        const float* __restrict__ v,
                        const float* __restrict__ ln_g, const float* __restrict__ ln_b,
                        float* __restrict__ x, _Float16* __restrict__ xh) {
    __shared__ float sc[N_][N_];
    __shared__ float xbuf[N_ * H_];
    __shared__ float ps[256], ps2[256];
    __shared__ float mu[N_], rs[N_];

    const int b = blockIdx.x;
    const int t = threadIdx.x;
    const int n = t >> 4, m = t & 15;

    // scores[n][m] = q_n . k_m / sqrt(H)
    {
        const float* qr = q + ((size_t)b * N_ + n) * H_;
        const float* kr = k + ((size_t)b * N_ + m) * H_;
        float acc = 0.f;
        for (int d = 0; d < H_; ++d) acc += qr[d] * kr[d];
        sc[n][m] = acc * 0.036084391824351615f;  // 1/sqrt(768)
    }
    __syncthreads();

    // row softmax (16 threads, one row each)
    if (t < N_) {
        float mx = sc[t][0];
        for (int j = 1; j < N_; ++j) mx = fmaxf(mx, sc[t][j]);
        float s = 0.f;
        for (int j = 0; j < N_; ++j) { float e = expf(sc[t][j] - mx); sc[t][j] = e; s += e; }
        float inv = 1.f / s;
        for (int j = 0; j < N_; ++j) sc[t][j] *= inv;
    }
    __syncthreads();

    // attn + residual into LDS
    for (int idx = t; idx < N_ * H_; idx += 256) {
        int nn = idx / H_, h = idx - nn * H_;
        float a2 = 0.f;
        #pragma unroll
        for (int mm = 0; mm < N_; ++mm)
            a2 += sc[nn][mm] * v[((size_t)b * N_ + mm) * H_ + h];
        xbuf[idx] = pooled[((size_t)b * N_ + nn) * H_ + h] + a2;
    }
    __syncthreads();

    // per-row mean / var (16 threads per row)
    {
        int row = t >> 4, l16 = t & 15;
        float s = 0.f, s2 = 0.f;
        for (int j = l16; j < H_; j += 16) {
            float xv = xbuf[row * H_ + j];
            s += xv; s2 += xv * xv;
        }
        ps[t] = s; ps2[t] = s2;
    }
    __syncthreads();
    if (t < N_) {
        float s = 0.f, s2 = 0.f;
        for (int j = 0; j < 16; ++j) { s += ps[t * 16 + j]; s2 += ps2[t * 16 + j]; }
        float mean = s / (float)H_;
        float var  = s2 / (float)H_ - mean * mean;
        mu[t] = mean;
        rs[t] = rsqrtf(var + 1e-5f);
    }
    __syncthreads();

    for (int idx = t; idx < N_ * H_; idx += 256) {
        int nn = idx / H_, h = idx - nn * H_;
        float xv = (xbuf[idx] - mu[nn]) * rs[nn] * ln_g[h] + ln_b[h];
        size_t o = ((size_t)b * N_ + nn) * H_ + h;
        x[o]  = xv;
        xh[o] = (_Float16)xv;
    }
}

// ---------------------------------------------------------------------------
// Component head: logits = hidden[row] @ w2 + b2 ; per-row CE.
// One block (64 threads) per row of 1024.
// ---------------------------------------------------------------------------
__global__ void comp_head(const float* __restrict__ hidden, const float* __restrict__ w2,
                          const float* __restrict__ b2, const int* __restrict__ gold,
                          float* __restrict__ outLogits, float* __restrict__ rowCE) {
    __shared__ float red[64 * CC_];
    const int row = blockIdx.x, t = threadIdx.x;
    float acc[CC_] = {0.f, 0.f, 0.f, 0.f, 0.f};
    const float* hr = hidden + (size_t)row * HID_;
    for (int e = t; e < HID_; e += 64) {
        float hv = hr[e];
        #pragma unroll
        for (int c = 0; c < CC_; ++c) acc[c] += hv * w2[e * CC_ + c];
    }
    #pragma unroll
    for (int c = 0; c < CC_; ++c) red[t * CC_ + c] = acc[c];
    __syncthreads();
    if (t == 0) {
        float lg[CC_];
        #pragma unroll
        for (int c = 0; c < CC_; ++c) {
            float s = 0.f;
            for (int j = 0; j < 64; ++j) s += red[j * CC_ + c];
            lg[c] = s + b2[c];
        }
        float mx = lg[0];
        #pragma unroll
        for (int c = 1; c < CC_; ++c) mx = fmaxf(mx, lg[c]);
        float se = 0.f;
        #pragma unroll
        for (int c = 0; c < CC_; ++c) se += expf(lg[c] - mx);
        float lse = logf(se) + mx;
        rowCE[row] = lse - lg[gold[row]];
        #pragma unroll
        for (int c = 0; c < CC_; ++c) outLogits[(size_t)row * CC_ + c] = lg[c];
    }
}

// ---------------------------------------------------------------------------
// Relation head for ordered pair p=(i,j): h = tanh(xt_i + xb_j) (b1 folded
// into xt), logits = h @ w2 + b2, per-pair CE. One block (128 thr) per pair.
// ---------------------------------------------------------------------------
__global__ void rel_head(const float* __restrict__ xt, const float* __restrict__ xb,
                         const float* __restrict__ w2, const float* __restrict__ b2,
                         const int* __restrict__ gold, float* __restrict__ outLogits,
                         float* __restrict__ rowCE) {
    __shared__ float red[128 * CR_];
    const int pidx = blockIdx.x;          // b*P_ + p
    const int b = pidx / P_;
    const int p = pidx - b * P_;
    const int i = p / 15;
    const int r = p - i * 15;
    const int j = r + (r >= i ? 1 : 0);
    const int t = threadIdx.x;

    const float* ti = xt + ((size_t)b * N_ + i) * HID_;
    const float* bj = xb + ((size_t)b * N_ + j) * HID_;
    float acc[CR_] = {0.f, 0.f, 0.f};
    for (int e = t; e < HID_; e += 128) {
        float hv = tanhf(ti[e] + bj[e]);
        #pragma unroll
        for (int c = 0; c < CR_; ++c) acc[c] += hv * w2[e * CR_ + c];
    }
    #pragma unroll
    for (int c = 0; c < CR_; ++c) red[t * CR_ + c] = acc[c];
    __syncthreads();
    if (t == 0) {
        float lg[CR_];
        #pragma unroll
        for (int c = 0; c < CR_; ++c) {
            float s = 0.f;
            for (int jj = 0; jj < 128; ++jj) s += red[jj * CR_ + c];
            lg[c] = s + b2[c];
        }
        float mx = fmaxf(lg[0], fmaxf(lg[1], lg[2]));
        float se = expf(lg[0] - mx) + expf(lg[1] - mx) + expf(lg[2] - mx);
        float lse = logf(se) + mx;
        rowCE[pidx] = lse - lg[gold[pidx]];
        #pragma unroll
        for (int c = 0; c < CR_; ++c) outLogits[(size_t)pidx * CR_ + c] = lg[c];
    }
}

// ---------------------------------------------------------------------------
// Deterministic mean reduction: block 0 -> loss_comp, block 1 -> loss_rel.
// ---------------------------------------------------------------------------
__global__ void loss_reduce(const float* __restrict__ ceC, const float* __restrict__ ceR,
                            float* __restrict__ out) {
    __shared__ float s[256];
    const int t = threadIdx.x;
    const float* src = (blockIdx.x == 0) ? ceC : ceR;
    const int n = (blockIdx.x == 0) ? (B_ * N_) : (B_ * P_);
    float acc = 0.f;
    for (int i = t; i < n; i += 256) acc += src[i];
    s[t] = acc;
    __syncthreads();
    for (int off = 128; off > 0; off >>= 1) {
        if (t < off) s[t] += s[t + off];
        __syncthreads();
    }
    if (t == 0) out[blockIdx.x] = s[0] / (float)n;
}

// ---------------------------------------------------------------------------
extern "C" void kernel_launch(void* const* d_in, const int* in_sizes, int n_in,
                              void* d_out, int out_size, void* d_ws, size_t ws_size,
                              hipStream_t stream) {
    const float* lhs      = (const float*)d_in[0];
    const int*   starts   = (const int*)d_in[1];
    const int*   lens     = (const int*)d_in[2];
    const int*   comps    = (const int*)d_in[3];
    const int*   rels     = (const int*)d_in[4];
    const float* comp_w1  = (const float*)d_in[5];
    const float* comp_b1  = (const float*)d_in[6];
    const float* comp_w2  = (const float*)d_in[7];
    const float* comp_b2  = (const float*)d_in[8];
    const float* wq       = (const float*)d_in[9];
    const float* wk       = (const float*)d_in[10];
    const float* wv       = (const float*)d_in[11];
    const float* ln_g     = (const float*)d_in[12];
    const float* ln_b     = (const float*)d_in[13];
    const float* rel_w1   = (const float*)d_in[14];
    const float* rel_b1   = (const float*)d_in[15];
    const float* rel_w2   = (const float*)d_in[16];
    const float* rel_b2   = (const float*)d_in[17];
    float* out = (float*)d_out;

    // ---- workspace bump allocator (256-B aligned slabs) ----
    char* ws = (char*)d_ws;
    size_t off = 0;
    auto alloc = [&](size_t bytes) -> void* {
        void* p = ws + off;
        off += (bytes + 255) & ~(size_t)255;
        return p;
    };
    const int M = B_ * N_;                       // 1024 rows
    float*     pooled  = (float*)alloc((size_t)M * H_ * 4);
    _Float16*  pooledH = (_Float16*)alloc((size_t)M * H_ * 2);
    float*     xbuf    = (float*)alloc((size_t)M * H_ * 4);
    _Float16*  xh      = (_Float16*)alloc((size_t)M * H_ * 2);
    float*     hidden  = (float*)alloc((size_t)M * HID_ * 4);
    float*     qb      = (float*)alloc((size_t)M * H_ * 4);
    float*     kb      = (float*)alloc((size_t)M * H_ * 4);
    float*     vb      = (float*)alloc((size_t)M * H_ * 4);
    float*     xt      = (float*)alloc((size_t)M * HID_ * 4);
    float*     xbp     = (float*)alloc((size_t)M * HID_ * 4);
    uint32_t*  cw1p    = (uint32_t*)alloc((size_t)(H_ / 2) * HID_ * 4);
    uint32_t*  wqp     = (uint32_t*)alloc((size_t)(H_ / 2) * H_ * 4);
    uint32_t*  wkp     = (uint32_t*)alloc((size_t)(H_ / 2) * H_ * 4);
    uint32_t*  wvp     = (uint32_t*)alloc((size_t)(H_ / 2) * H_ * 4);
    uint32_t*  rtp     = (uint32_t*)alloc((size_t)(H_ / 2) * HID_ * 4);
    uint32_t*  rbp     = (uint32_t*)alloc((size_t)(H_ / 2) * HID_ * 4);
    float*     ceC     = (float*)alloc((size_t)M * 4);
    float*     ceR     = (float*)alloc((size_t)B_ * P_ * 4);
    (void)ws_size; (void)in_sizes; (void)n_in; (void)out_size;

    // ---- pack weights into WMMA-native tiled f16 layout ----
    {
        int t512 = (HID_ / 16) * (H_ / 32) * 256;   // dwords: 32*24*256
        int t768 = (H_ / 16)  * (H_ / 32) * 256;    // dwords: 48*24*256
        pack_b_tiles<<<(t512 + 255) / 256, 256, 0, stream>>>(comp_w1, cw1p, HID_, H_, 0);
        pack_b_tiles<<<(t768 + 255) / 256, 256, 0, stream>>>(wq, wqp, H_, H_, 0);
        pack_b_tiles<<<(t768 + 255) / 256, 256, 0, stream>>>(wk, wkp, H_, H_, 0);
        pack_b_tiles<<<(t768 + 255) / 256, 256, 0, stream>>>(wv, wvp, H_, H_, 0);
        pack_b_tiles<<<(t512 + 255) / 256, 256, 0, stream>>>(rel_w1, rtp, HID_, H_, 0);
        pack_b_tiles<<<(t512 + 255) / 256, 256, 0, stream>>>(rel_w1, rbp, HID_, H_, H_);
    }

    // ---- span pooling ----
    pool_spans<<<M, 256, 0, stream>>>(lhs, starts, lens, pooled, pooledH);

    // ---- WMMA GEMMs on pooled (16x64 strip per wave) ----
    wmma_gemm4<<<dim3(M / 16, HID_ / 64), 32, 0, stream>>>(
        pooledH, cw1p, comp_b1, hidden, HID_, H_, /*tanh=*/1);
    wmma_gemm4<<<dim3(M / 16, H_ / 64), 32, 0, stream>>>(
        pooledH, wqp, nullptr, qb, H_, H_, 0);
    wmma_gemm4<<<dim3(M / 16, H_ / 64), 32, 0, stream>>>(
        pooledH, wkp, nullptr, kb, H_, H_, 0);
    wmma_gemm4<<<dim3(M / 16, H_ / 64), 32, 0, stream>>>(
        pooledH, wvp, nullptr, vb, H_, H_, 0);

    // ---- attention + residual + layernorm ----
    attn_ln<<<B_, 256, 0, stream>>>(pooled, qb, kb, vb, ln_g, ln_b, xbuf, xh);

    // ---- relation pre-projections (b1 folded into xt) ----
    wmma_gemm4<<<dim3(M / 16, HID_ / 64), 32, 0, stream>>>(
        xh, rtp, rel_b1, xt, HID_, H_, 0);
    wmma_gemm4<<<dim3(M / 16, HID_ / 64), 32, 0, stream>>>(
        xh, rbp, nullptr, xbp, HID_, H_, 0);

    // ---- heads + per-row CE ----
    comp_head<<<M, 64, 0, stream>>>(hidden, comp_w2, comp_b2, comps, out, ceC);
    rel_head<<<B_ * P_, 128, 0, stream>>>(xt, xbp, rel_w2, rel_b2, rels,
                                          out + (size_t)M * CC_, ceR);

    // ---- mean losses ----
    loss_reduce<<<2, 256, 0, stream>>>(ceC, ceR,
                                       out + (size_t)M * CC_ + (size_t)B_ * P_ * CR_);
}
